// GATNet_69097433858678
// MI455X (gfx1250) — compile-verified, matrix operand order checked by reading, not run on hardware
//
#include <hip/hip_runtime.h>
#include <hip/hip_bf16.h>
#include <math.h>

typedef __bf16 bf16_t;
typedef bf16_t v16bf __attribute__((ext_vector_type(16)));
typedef bf16_t v8bf  __attribute__((ext_vector_type(8)));
typedef float  v8f   __attribute__((ext_vector_type(8)));

#define TPB 256

static __device__ __forceinline__ float lrelu(float x) { return x > 0.f ? x : 0.2f * x; }

// float atomic max via sign-split int/uint atomics (works with -inf init)
static __device__ __forceinline__ void atomicMaxF(float* addr, float v) {
    if (v >= 0.f) atomicMax((int*)addr, __float_as_int(v));
    else          atomicMin((unsigned int*)addr, __float_as_uint(v));
}

__global__ void k_fill(float* __restrict__ p, float v, int n) {
    int i = blockIdx.x * blockDim.x + threadIdx.x;
    if (i < n) p[i] = v;
}

// W1[512,64] f32 -> w1t[64,512] bf16 (transposed, for contiguous B-fragment loads)
__global__ void k_prep_w1(const float* __restrict__ W1, bf16_t* __restrict__ w1t) {
    int idx = blockIdx.x * blockDim.x + threadIdx.x;
    if (idx >= 64 * 512) return;
    int c = idx >> 9, k = idx & 511;
    w1t[idx] = (bf16_t)W1[k * 64 + c];
}

// h1[N,64] = x[N,512] @ W1[512,64] via v_wmma_f32_16x16x32_bf16.
// One wave per 16-row tile; 4 accumulators cover 64 output columns.
__global__ void k_gemm1_wmma(const float* __restrict__ x,
                             const bf16_t* __restrict__ w1t,
                             float* __restrict__ h1, int n) {
    int wave = blockIdx.x * (blockDim.x >> 5) + (threadIdx.x >> 5);
    int lane = threadIdx.x & 31;
    int row0 = wave * 16;
    if (row0 >= n) return;                 // wave-uniform exit: EXEC stays all-ones for WMMA
    int r = lane & 15;
    int khalf = (lane >> 4) << 3;          // 0 for lanes 0-15, 8 for lanes 16-31

    v8f acc[4] = {};
    const float* xrow = x + (size_t)(row0 + r) * 512;
    for (int k0 = 0; k0 < 512; k0 += 32) {
        // A fragment: elements 0..7 -> K=k0+khalf+e ; 8..15 -> K=k0+16+khalf+(e-8)
        float4 a0 = *(const float4*)(xrow + k0 + khalf);
        float4 a1 = *(const float4*)(xrow + k0 + khalf + 4);
        float4 a2 = *(const float4*)(xrow + k0 + khalf + 16);
        float4 a3 = *(const float4*)(xrow + k0 + khalf + 20);
        v16bf a;
        a[0]=(bf16_t)a0.x; a[1]=(bf16_t)a0.y; a[2]=(bf16_t)a0.z; a[3]=(bf16_t)a0.w;
        a[4]=(bf16_t)a1.x; a[5]=(bf16_t)a1.y; a[6]=(bf16_t)a1.z; a[7]=(bf16_t)a1.w;
        a[8]=(bf16_t)a2.x; a[9]=(bf16_t)a2.y; a[10]=(bf16_t)a2.z; a[11]=(bf16_t)a2.w;
        a[12]=(bf16_t)a3.x; a[13]=(bf16_t)a3.y; a[14]=(bf16_t)a3.z; a[15]=(bf16_t)a3.w;
#pragma unroll
        for (int j = 0; j < 4; j++) {
            const bf16_t* wcol = w1t + (size_t)(16 * j + r) * 512;   // column of W1
            v8bf blo = *(const v8bf*)(wcol + k0 + khalf);
            v8bf bhi = *(const v8bf*)(wcol + k0 + khalf + 16);
            v16bf b;
#pragma unroll
            for (int i = 0; i < 8; i++) { b[i] = blo[i]; b[8 + i] = bhi[i]; }
            acc[j] = __builtin_amdgcn_wmma_f32_16x16x32_bf16(
                false, a, false, b, (short)0, acc[j], false, false);
        }
    }
    // C/D layout: lane holds column (lane&15); rows = vgpr + (lane>=16 ? 8 : 0)
    int mbase = row0 + ((lane >> 4) << 3);
#pragma unroll
    for (int j = 0; j < 4; j++)
#pragma unroll
        for (int i = 0; i < 8; i++)
            h1[(size_t)(mbase + i) * 64 + 16 * j + r] = acc[j][i];
}

// el1/er1[N,8]: per-head attention dots
__global__ void k_el_er_1(const float* __restrict__ h1,
                          const float* __restrict__ al1, const float* __restrict__ ar1,
                          float* __restrict__ el1, float* __restrict__ er1, int n) {
    int idx = blockIdx.x * blockDim.x + threadIdx.x;
    if (idx >= n * 8) return;
    int node = idx >> 3, head = idx & 7;
    const float* hp = h1 + (size_t)node * 64 + head * 8;
    float a = 0.f, b = 0.f;
#pragma unroll
    for (int d = 0; d < 8; d++) { float v = hp[d]; a += v * al1[head * 8 + d]; b += v * ar1[head * 8 + d]; }
    el1[idx] = a; er1[idx] = b;
}

__global__ void k_edge_max1(const int* __restrict__ src, const int* __restrict__ dst,
                            const float* __restrict__ el1, const float* __restrict__ er1,
                            float* __restrict__ emax1, int e) {
    int idx = blockIdx.x * blockDim.x + threadIdx.x;
    if (idx >= e * 8) return;
    int eid = idx >> 3, head = idx & 7;
    int s = src[eid], d = dst[eid];
    float v = lrelu(el1[s * 8 + head] + er1[d * 8 + head]);
    atomicMaxF(&emax1[d * 8 + head], v);
}

// unnormalized aggregation: acc1[dst] += ee * h1[src]; denom1[dst] += ee
__global__ void k_edge_acc1(const int* __restrict__ src, const int* __restrict__ dst,
                            const float* __restrict__ el1, const float* __restrict__ er1,
                            const float* __restrict__ emax1, const float* __restrict__ h1,
                            float* __restrict__ denom1, float* __restrict__ acc1, int e) {
    int idx = blockIdx.x * blockDim.x + threadIdx.x;
    if (idx >= e * 8) return;
    int eid = idx >> 3, head = idx & 7;
    int s = src[eid], d = dst[eid];
    float v = lrelu(el1[s * 8 + head] + er1[d * 8 + head]);
    float ee = expf(v - emax1[d * 8 + head]);
    atomicAdd(&denom1[d * 8 + head], ee);
    const float* hp = h1 + (size_t)s * 64 + head * 8;
    float* ap = acc1 + (size_t)d * 64 + head * 8;
#pragma unroll
    for (int i = 0; i < 8; i++) atomicAdd(&ap[i], ee * hp[i]);
}

// normalize + bias + ELU, in place on acc1 -> becomes layer-2 input
__global__ void k_finish1(float* __restrict__ acc1, const float* __restrict__ denom1,
                          const float* __restrict__ b1, int n) {
    int idx = blockIdx.x * blockDim.x + threadIdx.x;
    if (idx >= n * 64) return;
    int node = idx >> 6, col = idx & 63, head = col >> 3;
    float v = acc1[idx] / (denom1[node * 8 + head] + 1e-9f) + b1[col];
    acc1[idx] = v > 0.f ? v : expf(v) - 1.f;   // ELU(alpha=1)
}

// h2[N,40] = hact[N,64] @ W2[64,40]; fused el2/er2. W2 staged in LDS.
__global__ void k_gemm2(const float* __restrict__ hact, const float* __restrict__ W2,
                        const float* __restrict__ al2, const float* __restrict__ ar2,
                        float* __restrict__ h2, float* __restrict__ el2,
                        float* __restrict__ er2, int n) {
    __shared__ float w[64 * 40];
    for (int i = threadIdx.x; i < 64 * 40; i += blockDim.x) w[i] = W2[i];
    __syncthreads();
    int node = blockIdx.x * blockDim.x + threadIdx.x;
    if (node >= n) return;
    float xr[64];
    const float4* hp = (const float4*)(hact + (size_t)node * 64);
#pragma unroll
    for (int i = 0; i < 16; i++) {
        float4 t = hp[i];
        xr[4 * i] = t.x; xr[4 * i + 1] = t.y; xr[4 * i + 2] = t.z; xr[4 * i + 3] = t.w;
    }
    float elv = 0.f, erv = 0.f;
    for (int c = 0; c < 40; c++) {
        float s = 0.f;
#pragma unroll
        for (int k = 0; k < 64; k++) s += xr[k] * w[k * 40 + c];
        h2[(size_t)node * 40 + c] = s;
        elv += s * al2[c]; erv += s * ar2[c];
    }
    el2[node] = elv; er2[node] = erv;
}

__global__ void k_edge_max2(const int* __restrict__ src, const int* __restrict__ dst,
                            const float* __restrict__ el2, const float* __restrict__ er2,
                            float* __restrict__ emax2, int e) {
    int eid = blockIdx.x * blockDim.x + threadIdx.x;
    if (eid >= e) return;
    int s = src[eid], d = dst[eid];
    atomicMaxF(&emax2[d], lrelu(el2[s] + er2[d]));
}

__global__ void k_edge_acc2(const int* __restrict__ src, const int* __restrict__ dst,
                            const float* __restrict__ el2, const float* __restrict__ er2,
                            const float* __restrict__ emax2, const float* __restrict__ h2,
                            float* __restrict__ denom2, float* __restrict__ acc2, int e) {
    int eid = blockIdx.x * blockDim.x + threadIdx.x;
    if (eid >= e) return;
    int s = src[eid], d = dst[eid];
    float ee = expf(lrelu(el2[s] + er2[d]) - emax2[d]);
    atomicAdd(&denom2[d], ee);
    const float* hp = h2 + (size_t)s * 40;
    float* ap = acc2 + (size_t)d * 40;
#pragma unroll
    for (int i = 0; i < 40; i++) atomicAdd(&ap[i], ee * hp[i]);
}

// normalize + bias (mean over H2=1 heads is identity) + log_softmax
__global__ void k_final(const float* __restrict__ acc2, const float* __restrict__ denom2,
                        const float* __restrict__ b2, float* __restrict__ out, int n) {
    int node = blockIdx.x * blockDim.x + threadIdx.x;
    if (node >= n) return;
    float inv = 1.f / (denom2[node] + 1e-9f);
    const float* ap = acc2 + (size_t)node * 40;
    float v[40], m = -INFINITY;
#pragma unroll
    for (int c = 0; c < 40; c++) { v[c] = ap[c] * inv + b2[c]; m = fmaxf(m, v[c]); }
    float s = 0.f;
#pragma unroll
    for (int c = 0; c < 40; c++) s += expf(v[c] - m);
    float ls = m + logf(s);
    float* op = out + (size_t)node * 40;
#pragma unroll
    for (int c = 0; c < 40; c++) op[c] = v[c] - ls;
}

extern "C" void kernel_launch(void* const* d_in, const int* in_sizes, int n_in,
                              void* d_out, int out_size, void* d_ws, size_t ws_size,
                              hipStream_t stream) {
    const float* x   = (const float*)d_in[0];
    const int*   src = (const int*)  d_in[1];
    const int*   dst = (const int*)  d_in[2];
    const float* W1  = (const float*)d_in[3];
    const float* al1 = (const float*)d_in[4];
    const float* ar1 = (const float*)d_in[5];
    const float* b1  = (const float*)d_in[6];
    const float* W2  = (const float*)d_in[7];
    const float* al2 = (const float*)d_in[8];
    const float* ar2 = (const float*)d_in[9];
    const float* b2  = (const float*)d_in[10];
    float* out = (float*)d_out;
    int n = in_sizes[0] / 512;
    int e = in_sizes[1];

    // bump allocator over workspace (256B aligned)
    char* p = (char*)d_ws;
    auto alloc = [&](size_t bytes) -> void* {
        void* q = (void*)p; p += (bytes + 255) & ~(size_t)255; return q;
    };
    bf16_t* w1t   = (bf16_t*)alloc((size_t)64 * 512 * sizeof(bf16_t));
    float* h1     = (float*)alloc((size_t)n * 64 * 4);
    float* el1    = (float*)alloc((size_t)n * 8 * 4);
    float* er1    = (float*)alloc((size_t)n * 8 * 4);
    float* emax1  = (float*)alloc((size_t)n * 8 * 4);
    float* denom1 = (float*)alloc((size_t)n * 8 * 4);
    float* acc1   = (float*)alloc((size_t)n * 64 * 4);
    float* h2     = (float*)alloc((size_t)n * 40 * 4);
    float* el2    = (float*)alloc((size_t)n * 4);
    float* er2    = (float*)alloc((size_t)n * 4);
    float* emax2  = (float*)alloc((size_t)n * 4);
    float* denom2 = (float*)alloc((size_t)n * 4);
    float* acc2   = (float*)alloc((size_t)n * 40 * 4);

    auto blk = [](long long t) { return (int)((t + TPB - 1) / TPB); };

    // ---- layer 1 ----
    k_prep_w1<<<blk(64 * 512), TPB, 0, stream>>>(W1, w1t);
    int waves = (n + 15) / 16;
    k_gemm1_wmma<<<(waves + 7) / 8, TPB, 0, stream>>>(x, w1t, h1, n);
    k_el_er_1<<<blk((long long)n * 8), TPB, 0, stream>>>(h1, al1, ar1, el1, er1, n);
    k_fill<<<blk((long long)n * 8), TPB, 0, stream>>>(emax1, -INFINITY, n * 8);
    k_fill<<<blk((long long)n * 8), TPB, 0, stream>>>(denom1, 0.f, n * 8);
    k_fill<<<blk((long long)n * 64), TPB, 0, stream>>>(acc1, 0.f, n * 64);
    k_edge_max1<<<blk((long long)e * 8), TPB, 0, stream>>>(src, dst, el1, er1, emax1, e);
    k_edge_acc1<<<blk((long long)e * 8), TPB, 0, stream>>>(src, dst, el1, er1, emax1, h1,
                                                           denom1, acc1, e);
    k_finish1<<<blk((long long)n * 64), TPB, 0, stream>>>(acc1, denom1, b1, n);

    // ---- layer 2 ----
    k_gemm2<<<blk(n), TPB, 0, stream>>>(acc1, W2, al2, ar2, h2, el2, er2, n);
    k_fill<<<blk(n), TPB, 0, stream>>>(emax2, -INFINITY, n);
    k_fill<<<blk(n), TPB, 0, stream>>>(denom2, 0.f, n);
    k_fill<<<blk((long long)n * 40), TPB, 0, stream>>>(acc2, 0.f, n * 40);
    k_edge_max2<<<blk(e), TPB, 0, stream>>>(src, dst, el2, er2, emax2, e);
    k_edge_acc2<<<blk(e), TPB, 0, stream>>>(src, dst, el2, er2, emax2, h2, denom2, acc2, e);
    k_final<<<blk(n), TPB, 0, stream>>>(acc2, denom2, b2, out, n);
}
